// MambaForecaster_19748259627345
// MI455X (gfx1250) — matrix operand order repaired and checked by reading, not compiled, not run on hardware
//
#include <hip/hip_runtime.h>
#include <math.h>

// ---------------------------------------------------------------------------
// Problem constants (from reference)
// ---------------------------------------------------------------------------
#define B_    8
#define SEQ_  4096
#define CIN_  7
#define DM_   512
#define NL_   4
#define PL_   16
#define NP_   256          // SEQ / PL
#define NS_   16
#define DC_   4
#define DI_   1024         // EXP * DM
#define DTR_  32
#define CD_   64
#define PRED_ 96
#define NTOK  (B_ * NP_)   // 2048 tokens
#define HFLAT (NP_ * DM_)  // 131072

typedef __bf16 bf16_t;
typedef __attribute__((ext_vector_type(16))) __bf16 v16bf;
typedef __attribute__((ext_vector_type(8)))  __bf16 v8bf;
typedef __attribute__((ext_vector_type(8)))  float  v8f;

union V16U { v16bf v; v8bf h[2]; };

// ---------------------------------------------------------------------------
// fp32 -> bf16 conversion
// ---------------------------------------------------------------------------
__global__ void f2bf_kernel(const float* __restrict__ src, bf16_t* __restrict__ dst, int n) {
    int i = blockIdx.x * 256 + threadIdx.x;
    if (i < n) dst[i] = (bf16_t)src[i];
}

// ---------------------------------------------------------------------------
// Generic WMMA bf16 GEMM:  C[M,N] (+)= A[M,K] * W[N,K]^T   (fp32 accum)
// block = 128 threads (4 waves); block tile 64 x (32*WN)
// wave tile 32 x (16*WN): WN=4 -> 8 WMMAs / K-step (main GEMMs), WN=2 for N=64
// ---------------------------------------------------------------------------
template<int WN>
__global__ __launch_bounds__(128)
void gemm_bf16_wmma(const bf16_t* __restrict__ A, const bf16_t* __restrict__ W,
                    float* __restrict__ C, int K,
                    int lda, int ldb, int ldc, int accum) {
    const int lane = threadIdx.x & 31;
    const int wave = threadIdx.x >> 5;
    const int m0 = blockIdx.x * 64 + (wave & 1) * 32;
    const int n0 = blockIdx.y * (32 * WN) + (wave >> 1) * (16 * WN);
    const int r   = lane & 15;      // row within 16-tile
    const int sel = lane >> 4;      // lane-group select (K swizzle)

    const bf16_t* pa0 = A + (size_t)(m0 + r) * lda;
    const bf16_t* pa1 = pa0 + (size_t)16 * lda;
    const bf16_t* pb[WN];
    #pragma unroll
    for (int j = 0; j < WN; ++j)
        pb[j] = W + (size_t)(n0 + j * 16 + r) * ldb + sel * 16;

    v8f cl[WN], ch[WN];
    #pragma unroll
    for (int j = 0; j < WN; ++j) { cl[j] = (v8f){}; ch[j] = (v8f){}; }

    for (int k = 0; k < K; k += 32) {
        // A-frag (16x32, MxK): lanes 0-15 hold K[0..15], lanes 16-31 K[16..31]
        V16U a0, a1;
        a0.h[0] = *(const v8bf*)(pa0 + k + sel * 16);
        a0.h[1] = *(const v8bf*)(pa0 + k + sel * 16 + 8);
        a1.h[0] = *(const v8bf*)(pa1 + k + sel * 16);
        a1.h[1] = *(const v8bf*)(pa1 + k + sel * 16 + 8);
        // prefetch one K-block (128 elems) ahead (global_prefetch_b8)
        __builtin_prefetch(pa0 + k + 128, 0, 1);
        __builtin_prefetch(pb[0] + k + 128, 0, 1);
        #pragma unroll
        for (int j = 0; j < WN; ++j) {
            // B-frag (32x16, KxN): lane n holds 16 contiguous K starting at sel*16 (row n of W)
            v16bf bj = *(const v16bf*)(pb[j] + k);
            cl[j] = __builtin_amdgcn_wmma_f32_16x16x32_bf16(false, a0.v, false, bj, (short)0, cl[j], false, false);
            ch[j] = __builtin_amdgcn_wmma_f32_16x16x32_bf16(false, a1.v, false, bj, (short)0, ch[j], false, false);
        }
    }

    // C/D layout: VGPR q -> row (m + sel*8 + q), col (n + lane&15)
    const int col   = lane & 15;
    const int rbase = sel * 8;
    #pragma unroll
    for (int j = 0; j < WN; ++j) {
        float* pl = C + (size_t)(m0 + rbase) * ldc + (n0 + j * 16 + col);
        float* ph = pl + (size_t)16 * ldc;
        #pragma unroll
        for (int q = 0; q < 8; ++q) {
            float vl = cl[j][q], vh = ch[j][q];
            if (accum) { vl += pl[(size_t)q * ldc]; vh += ph[(size_t)q * ldc]; }
            pl[(size_t)q * ldc] = vl;
            ph[(size_t)q * ldc] = vh;
        }
    }
}

// ---------------------------------------------------------------------------
// Input layernorm (over CIN=7) + patch embedding -> h[NTOK, DM] fp32
// one block per token, 512 threads
// ---------------------------------------------------------------------------
__global__ __launch_bounds__(512)
void patch_embed_kernel(const float* __restrict__ x, const float* __restrict__ ing,
                        const float* __restrict__ inb, const float* __restrict__ peW,
                        const float* __restrict__ peb, float* __restrict__ h) {
    __shared__ float lp[CIN_ * PL_];          // [c*16 + p]
    const int tok = blockIdx.x;
    const int b  = tok >> 8;
    const int np = tok & 255;
    const int tid = threadIdx.x;
    if (tid < PL_) {
        const float* row = x + ((size_t)b * SEQ_ + (size_t)np * PL_ + tid) * CIN_;
        float m = 0.f;
        for (int c = 0; c < CIN_; ++c) m += row[c];
        m *= (1.0f / CIN_);
        float v = 0.f;
        for (int c = 0; c < CIN_; ++c) { float d = row[c] - m; v += d * d; }
        v *= (1.0f / CIN_);
        float inv = rsqrtf(v + 1e-5f);
        for (int c = 0; c < CIN_; ++c)
            lp[c * PL_ + tid] = (row[c] - m) * inv * ing[c] + inb[c];
    }
    __syncthreads();
    float acc = peb[tid];
    const float* w = peW + (size_t)tid * (CIN_ * PL_);
    #pragma unroll 8
    for (int i = 0; i < CIN_ * PL_; ++i) acc += lp[i] * w[i];
    h[(size_t)tok * DM_ + tid] = acc;
}

// ---------------------------------------------------------------------------
// Token layernorm (over DM=512): optional bf16 and/or fp32 outputs
// one block per token, 256 threads (2 elems/thread)
// ---------------------------------------------------------------------------
__global__ __launch_bounds__(256)
void ln_token_kernel(const float* __restrict__ h, const float* __restrict__ g,
                     const float* __restrict__ bta, bf16_t* obf, float* of) {
    __shared__ float s1[256], s2[256];
    const int t = blockIdx.x;
    const int tid = threadIdx.x;
    const float* row = h + (size_t)t * DM_;
    float x0 = row[tid], x1 = row[tid + 256];
    s1[tid] = x0 + x1;
    s2[tid] = x0 * x0 + x1 * x1;
    __syncthreads();
    for (int s = 128; s > 0; s >>= 1) {
        if (tid < s) { s1[tid] += s1[tid + s]; s2[tid] += s2[tid + s]; }
        __syncthreads();
    }
    float m   = s1[0] * (1.0f / DM_);
    float var = s2[0] * (1.0f / DM_) - m * m;
    float inv = rsqrtf(var + 1e-5f);
    float y0 = (x0 - m) * inv * g[tid]       + bta[tid];
    float y1 = (x1 - m) * inv * g[tid + 256] + bta[tid + 256];
    if (obf) { obf[(size_t)t * DM_ + tid] = (bf16_t)y0; obf[(size_t)t * DM_ + tid + 256] = (bf16_t)y1; }
    if (of)  { of[(size_t)t * DM_ + tid] = y0;          of[(size_t)t * DM_ + tid + 256] = y1; }
}

// ---------------------------------------------------------------------------
// Depthwise causal conv (DC=4) over L + bias + SiLU. xi = xz[:, :DI]
// one thread per (token, d)
// ---------------------------------------------------------------------------
__global__ void conv_silu_kernel(const float* __restrict__ xz, const float* __restrict__ cW,
                                 const float* __restrict__ cb, float* __restrict__ xcf,
                                 bf16_t* __restrict__ xcb) {
    int idx = blockIdx.x * 256 + threadIdx.x;
    if (idx >= NTOK * DI_) return;
    int d   = idx & (DI_ - 1);
    int tok = idx >> 10;
    int b = tok >> 8;
    int l = tok & 255;
    const float* w = cW + (size_t)d * DC_;
    float acc = cb[d];
    #pragma unroll
    for (int j = 0; j < DC_; ++j) {
        int ls = l - (DC_ - 1) + j;
        if (ls >= 0) acc += xz[((size_t)(b * NP_ + ls)) * (2 * DI_) + d] * w[j];
    }
    float s = acc / (1.0f + expf(-acc));   // silu
    xcf[idx] = s;
    xcb[idx] = (bf16_t)s;
}

// ---------------------------------------------------------------------------
// delta = softplus(delta_raw + b_dt[d])  (in-place)
// ---------------------------------------------------------------------------
__global__ void softplus_bias_kernel(float* __restrict__ delta, const float* __restrict__ bdt) {
    int idx = blockIdx.x * 256 + threadIdx.x;
    if (idx >= NTOK * DI_) return;
    float v = delta[idx] + bdt[idx & (DI_ - 1)];
    delta[idx] = (v > 20.0f) ? v : log1pf(expf(v));
}

// ---------------------------------------------------------------------------
// Selective scan: one thread per (b, d); 16 states in registers, 256 steps
// dbl row layout: [0:32)=dt (unused here), [32:48)=B, [48:64)=C
// ---------------------------------------------------------------------------
__global__ __launch_bounds__(256)
void scan_kernel(const float* __restrict__ delta, const float* __restrict__ xc,
                 const float* __restrict__ dbl, const float* __restrict__ A_log,
                 float* __restrict__ y) {
    int gidx = blockIdx.x * 256 + threadIdx.x;   // [0, B_*DI_)
    int b = gidx >> 10;
    int d = gidx & (DI_ - 1);
    float a[NS_], hst[NS_];
    #pragma unroll
    for (int n = 0; n < NS_; ++n) { a[n] = -expf(A_log[(size_t)d * NS_ + n]); hst[n] = 0.0f; }
    for (int t = 0; t < NP_; ++t) {
        size_t tok = (size_t)b * NP_ + t;
        float dl = delta[tok * DI_ + d];
        float xv = xc[tok * DI_ + d];
        const float* rowd = dbl + tok * 64;
        float yv = 0.0f;
        #pragma unroll
        for (int n = 0; n < NS_; ++n) {
            float dA = expf(dl * a[n]);
            hst[n] = dA * hst[n] + (dl * rowd[32 + n]) * xv;
            yv += hst[n] * rowd[48 + n];
        }
        y[tok * DI_ + d] = yv;
    }
}

// ---------------------------------------------------------------------------
// y2 = (y + xc * Dskip[d]) * silu(z),  z = xz[:, DI + d]  ->  bf16
// ---------------------------------------------------------------------------
__global__ void gate_kernel(const float* __restrict__ y, const float* __restrict__ xc,
                            const float* __restrict__ xz, const float* __restrict__ Dv,
                            bf16_t* __restrict__ ybf) {
    int idx = blockIdx.x * 256 + threadIdx.x;
    if (idx >= NTOK * DI_) return;
    int d   = idx & (DI_ - 1);
    int tok = idx >> 10;
    float z  = xz[(size_t)tok * (2 * DI_) + DI_ + d];
    float sz = z / (1.0f + expf(-z));
    ybf[idx] = (bf16_t)((y[idx] + xc[idx] * Dv[d]) * sz);
}

// ---------------------------------------------------------------------------
// Head layer 1: t1[b,j] = gelu(dot(hn[b,:], hW1[j,:]) + hb1[j]), dot len 131072
// grid (128, 8), 256-thread block reduction
// ---------------------------------------------------------------------------
__global__ __launch_bounds__(256)
void head1_kernel(const float* __restrict__ hn, const float* __restrict__ hW1,
                  const float* __restrict__ hb1, float* __restrict__ t1) {
    __shared__ float s[256];
    const int j = blockIdx.x, b = blockIdx.y;
    const float* hr = hn + (size_t)b * HFLAT;
    const float* wr = hW1 + (size_t)j * HFLAT;
    float acc = 0.0f;
    for (int i = threadIdx.x; i < HFLAT; i += 256) acc += hr[i] * wr[i];
    s[threadIdx.x] = acc;
    __syncthreads();
    for (int st = 128; st > 0; st >>= 1) {
        if (threadIdx.x < st) s[threadIdx.x] += s[threadIdx.x + st];
        __syncthreads();
    }
    if (threadIdx.x == 0) {
        float r = s[0] + hb1[j];
        r = 0.5f * r * (1.0f + erff(r * 0.70710678118654752f));   // exact gelu
        t1[b * (2 * CD_) + j] = r;
    }
}

// ---------------------------------------------------------------------------
// Head layer 2: out[b,o] = dot(t1[b,:], hW2[o,:]) + hb2[o]
// ---------------------------------------------------------------------------
__global__ __launch_bounds__(128)
void head2_kernel(const float* __restrict__ t1, const float* __restrict__ hW2,
                  const float* __restrict__ hb2, float* __restrict__ out) {
    const int b = blockIdx.x, o = threadIdx.x;
    if (o < PRED_) {
        float acc = hb2[o];
        #pragma unroll 8
        for (int k = 0; k < 2 * CD_; ++k) acc += t1[b * (2 * CD_) + k] * hW2[o * (2 * CD_) + k];
        out[b * PRED_ + o] = acc;
    }
}

// ---------------------------------------------------------------------------
// Launcher
// ---------------------------------------------------------------------------
extern "C" void kernel_launch(void* const* d_in, const int* in_sizes, int n_in,
                              void* d_out, int out_size, void* d_ws, size_t ws_size,
                              hipStream_t stream) {
    (void)in_sizes; (void)n_in; (void)out_size; (void)ws_size;

    const float* x      = (const float*)d_in[0];
    const float* in_g   = (const float*)d_in[1];
    const float* in_b   = (const float*)d_in[2];
    const float* pe_W   = (const float*)d_in[3];
    const float* pe_b   = (const float*)d_in[4];
    const float* ln_g   = (const float*)d_in[5];
    const float* ln_b   = (const float*)d_in[6];
    const float* W_in   = (const float*)d_in[7];
    const float* conv_W = (const float*)d_in[8];
    const float* conv_b = (const float*)d_in[9];
    const float* W_x    = (const float*)d_in[10];
    const float* W_dt   = (const float*)d_in[11];
    const float* b_dt   = (const float*)d_in[12];
    const float* A_log  = (const float*)d_in[13];
    const float* Dskip  = (const float*)d_in[14];
    const float* W_out  = (const float*)d_in[15];
    const float* fn_g   = (const float*)d_in[16];
    const float* fn_b   = (const float*)d_in[17];
    const float* hW1    = (const float*)d_in[18];
    const float* hb1    = (const float*)d_in[19];
    const float* hW2    = (const float*)d_in[20];
    const float* hb2    = (const float*)d_in[21];
    float* out = (float*)d_out;

    // --- workspace carving (256B aligned) ---
    char* ws = (char*)d_ws;
    size_t off = 0;
    auto alloc = [&](size_t bytes) -> void* {
        void* p = ws + off;
        off += (bytes + 255) & ~(size_t)255;
        return p;
    };
    float*  hbuf    = (float*) alloc((size_t)NTOK * DM_ * 4);          // residual stream
    bf16_t* u_bf    = (bf16_t*)alloc((size_t)NTOK * DM_ * 2);          // LN(h) bf16
    float*  xz      = (float*) alloc((size_t)NTOK * 2 * DI_ * 4);
    float*  xcf     = (float*) alloc((size_t)NTOK * DI_ * 4);
    bf16_t* xcb     = (bf16_t*)alloc((size_t)NTOK * DI_ * 2);
    float*  dbl     = (float*) alloc((size_t)NTOK * 64 * 4);
    bf16_t* dbl_bf  = (bf16_t*)alloc((size_t)NTOK * 64 * 2);
    float*  delta   = (float*) alloc((size_t)NTOK * DI_ * 4);
    float*  ybuf    = (float*) alloc((size_t)NTOK * DI_ * 4);
    bf16_t* y_bf    = (bf16_t*)alloc((size_t)NTOK * DI_ * 2);
    float*  hn      = (float*) alloc((size_t)B_ * HFLAT * 4);
    float*  t1      = (float*) alloc((size_t)B_ * 2 * CD_ * 4);
    bf16_t* W_in_bf  = (bf16_t*)alloc((size_t)NL_ * 2 * DI_ * DM_ * 2);
    bf16_t* W_x_bf   = (bf16_t*)alloc((size_t)NL_ * 64 * DI_ * 2);
    bf16_t* W_dt_bf  = (bf16_t*)alloc((size_t)NL_ * DI_ * DTR_ * 2);
    bf16_t* W_out_bf = (bf16_t*)alloc((size_t)NL_ * DM_ * DI_ * 2);

    // --- weight conversion fp32 -> bf16 ---
    auto cvt = [&](const float* s, bf16_t* d, int n) {
        f2bf_kernel<<<(n + 255) / 256, 256, 0, stream>>>(s, d, n);
    };
    cvt(W_in,  W_in_bf,  NL_ * 2 * DI_ * DM_);
    cvt(W_x,   W_x_bf,   NL_ * 64 * DI_);
    cvt(W_dt,  W_dt_bf,  NL_ * DI_ * DTR_);
    cvt(W_out, W_out_bf, NL_ * DM_ * DI_);

    // --- input LN + patch embed ---
    patch_embed_kernel<<<NTOK, 512, 0, stream>>>(x, in_g, in_b, pe_W, pe_b, hbuf);

    const int NELEM = NTOK * DI_;                    // 2,097,152
    const int EG = (NELEM + 255) / 256;              // elementwise grid

    for (int i = 0; i < NL_; ++i) {
        // LN -> bf16
        ln_token_kernel<<<NTOK, 256, 0, stream>>>(hbuf, ln_g + i * DM_, ln_b + i * DM_,
                                                  u_bf, (float*)nullptr);
        // xz = u @ W_in^T  (2048 x 2048 x 512), block tile 64x128
        gemm_bf16_wmma<4><<<dim3(NTOK / 64, (2 * DI_) / 128), 128, 0, stream>>>(
            u_bf, W_in_bf + (size_t)i * 2 * DI_ * DM_, xz,
            DM_, DM_, DM_, 2 * DI_, 0);
        // depthwise causal conv + silu
        conv_silu_kernel<<<EG, 256, 0, stream>>>(xz, conv_W + (size_t)i * DI_ * DC_,
                                                 conv_b + (size_t)i * DI_, xcf, xcb);
        // dbl = xc @ W_x^T  (2048 x 64 x 1024), block tile 64x64
        gemm_bf16_wmma<2><<<dim3(NTOK / 64, 1), 128, 0, stream>>>(
            xcb, W_x_bf + (size_t)i * 64 * DI_, dbl,
            DI_, DI_, DI_, 64, 0);
        // dt slice to bf16 (convert whole dbl; dt is columns [0,32) used via lda=64)
        f2bf_kernel<<<(NTOK * 64 + 255) / 256, 256, 0, stream>>>(dbl, dbl_bf, NTOK * 64);
        // delta_raw = dt @ W_dt^T  (2048 x 1024 x 32), block tile 64x128
        gemm_bf16_wmma<4><<<dim3(NTOK / 64, DI_ / 128), 128, 0, stream>>>(
            dbl_bf, W_dt_bf + (size_t)i * DI_ * DTR_, delta,
            DTR_, 64, DTR_, DI_, 0);
        // softplus(+ b_dt)
        softplus_bias_kernel<<<EG, 256, 0, stream>>>(delta, b_dt + (size_t)i * DI_);
        // selective scan
        scan_kernel<<<(B_ * DI_) / 256, 256, 0, stream>>>(delta, xcf, dbl,
                                                          A_log + (size_t)i * DI_ * NS_, ybuf);
        // gate: (y + xc*D) * silu(z) -> bf16
        gate_kernel<<<EG, 256, 0, stream>>>(ybuf, xcf, xz, Dskip + (size_t)i * DI_, y_bf);
        // h += y2 @ W_out^T  (2048 x 512 x 1024), accumulate residual, block tile 64x128
        gemm_bf16_wmma<4><<<dim3(NTOK / 64, DM_ / 128), 128, 0, stream>>>(
            y_bf, W_out_bf + (size_t)i * DM_ * DI_, hbuf,
            DI_, DI_, DI_, DM_, 1);
    }

    // final LN -> fp32 flat (B, NP*DM)
    ln_token_kernel<<<NTOK, 256, 0, stream>>>(hbuf, fn_g, fn_b, (bf16_t*)nullptr, hn);
    // head
    head1_kernel<<<dim3(2 * CD_, B_), 256, 0, stream>>>(hn, hW1, hb1, t1);
    head2_kernel<<<B_, 128, 0, stream>>>(t1, hW2, hb2, out);
}
